// GraphAttentionLayer_17171279249897
// MI455X (gfx1250) — matrix operand order repaired and checked by reading, not compiled, not run on hardware
//
#include <hip/hip_runtime.h>
#include <hip/hip_bf16.h>
#include <math.h>

#define N_NODES 8192
#define F_IN    512
#define F_OUT   256
#define HALF_N  4096
#define ALPHA_LR 0.2f

typedef __attribute__((ext_vector_type(16))) __bf16 v16bf;
typedef __attribute__((ext_vector_type(8)))  float  v8f;

union Frag16 { v16bf v; __bf16 h[16]; uint4 q[2]; };

static __device__ __forceinline__ v8f wmma_bf16(v16bf a, v16bf b, v8f c) {
  // D = A(16x32 bf16) * B(32x16 bf16) + C(16x16 f32)
  return __builtin_amdgcn_wmma_f32_16x16x32_bf16(false, a, false, b, (short)0, c,
                                                 false, false);
}

static __device__ __forceinline__ float lrelu(float x) { return x > 0.f ? x : ALPHA_LR * x; }
static __device__ __forceinline__ float elu1(float x)  { return x > 0.f ? x : expm1f(x); }

// ---------------------------------------------------------------- K0: WT = W^T (bf16)
__global__ void k_transpose_w(const float* __restrict__ W, __bf16* __restrict__ WT) {
  int tid = blockIdx.x * blockDim.x + threadIdx.x; // 0 .. F_IN*F_OUT-1
  int k = tid >> 8;       // row of W
  int c = tid & 255;      // col of W
  WT[(size_t)c * F_IN + k] = (__bf16)W[(size_t)k * F_OUT + c];
}

// ---------------------------------------------------------------- K1: hT = (x @ W)^T, bf16, via WMMA
// Workgroup: 256 thr (8 waves), 16 output rows, 256 cols (2 col-tiles/wave), K loop over 512.
__global__ void __launch_bounds__(256) k_gemm_h(const float* __restrict__ x,
                                                const __bf16* __restrict__ WT,
                                                __bf16* __restrict__ hT) {
  __shared__ __bf16 tileA[16 * 32];
  const int tid  = threadIdx.x;
  const int lane = tid & 31;
  const int wave = tid >> 5;
  const int hi   = lane >> 4;      // lane half selects K sub-range
  const int lcol = lane & 15;
  const int rowbase = blockIdx.x * 16;

  const int col0 = wave * 32 + lcol;
  const int col1 = col0 + 16;
  v8f acc0 = {}; v8f acc1 = {};

  for (int kb = 0; kb < F_IN; kb += 32) {
    __syncthreads();
    { // cooperative stage of x tile (f32 -> bf16)
      int r  = tid >> 4;
      int c2 = (tid & 15) * 2;
      const float2 vv = *reinterpret_cast<const float2*>(x + (size_t)(rowbase + r) * F_IN + kb + c2);
      tileA[r * 32 + c2]     = (__bf16)vv.x;
      tileA[r * 32 + c2 + 1] = (__bf16)vv.y;
    }
    __syncthreads();
    Frag16 a; // A layout: lanes 0-15 K 0-15, lanes 16-31 K 16-31
    {
      const __bf16* rp = &tileA[lcol * 32 + hi * 16];
      a.q[0] = *reinterpret_cast<const uint4*>(rp);
      a.q[1] = *reinterpret_cast<const uint4*>(rp + 8);
    }
    const int kc = kb + hi * 16; // B layout: lanes 0-15 K 0-15, lanes 16-31 K 16-31
    Frag16 b0, b1;
    { const uint4* pp = reinterpret_cast<const uint4*>(WT + (size_t)col0 * F_IN + kc); b0.q[0]=pp[0]; b0.q[1]=pp[1]; }
    { const uint4* pp = reinterpret_cast<const uint4*>(WT + (size_t)col1 * F_IN + kc); b1.q[0]=pp[0]; b1.q[1]=pp[1]; }
    acc0 = wmma_bf16(a.v, b0.v, acc0);
    acc1 = wmma_bf16(a.v, b1.v, acc1);
  }
  // C/D layout: element v -> row = v + 8*hi, col = lcol.  Write K-major hT (one b128 per tile).
  const int r0 = rowbase + hi * 8;
  union { uint4 q; __bf16 h[8]; } st;
#pragma unroll
  for (int v = 0; v < 8; ++v) st.h[v] = (__bf16)acc0[v];
  *reinterpret_cast<uint4*>(hT + (size_t)col0 * N_NODES + r0) = st.q;
#pragma unroll
  for (int v = 0; v < 8; ++v) st.h[v] = (__bf16)acc1[v];
  *reinterpret_cast<uint4*>(hT + (size_t)col1 * N_NODES + r0) = st.q;
}

// ---------------------------------------------------------------- K2: s = h@a1, t = h@a2
__global__ void k_st(const __bf16* __restrict__ hT, const float* __restrict__ a,
                     float* __restrict__ s, float* __restrict__ t) {
  int j = blockIdx.x * blockDim.x + threadIdx.x;
  float ss = 0.f, tt = 0.f;
  for (int c = 0; c < F_OUT; ++c) {
    float v = (float)hT[(size_t)c * N_NODES + j];
    ss += v * a[c];
    tt += v * a[F_OUT + c];
  }
  s[j] = ss; t[j] = tt;
}

// ---------------------------------------------------------------- K3: top-row logits, bottom p[] with global max
__global__ void __launch_bounds__(1024) k_scalars(const float* __restrict__ s, const float* __restrict__ t,
                                                  float* __restrict__ atop, float* __restrict__ btop,
                                                  float* __restrict__ p) {
  __shared__ float red[1024];
  const int tid = threadIdx.x;
  for (int i = tid; i < HALF_N; i += 1024) {
    atop[i] = lrelu(s[2 * i]     + t[2 * i]);
    btop[i] = lrelu(s[2 * i + 1] + t[2 * i + 1]);
  }
  float m = -3.0e38f;
  for (int j = tid; j < N_NODES; j += 1024) {
    float b = lrelu(s[(2 * j) & (N_NODES - 1)] + t[(2 * j + 1) & (N_NODES - 1)]);
    m = fmaxf(m, b);
  }
  red[tid] = m;
  __syncthreads();
  for (int off = 512; off > 0; off >>= 1) {
    if (tid < off) red[tid] = fmaxf(red[tid], red[tid + off]);
    __syncthreads();
  }
  const float M = red[0];
  for (int j = tid; j < N_NODES; j += 1024) {
    float b = lrelu(s[(2 * j) & (N_NODES - 1)] + t[(2 * j + 1) & (N_NODES - 1)]);
    p[j] = __expf(b - M);
  }
}

// ---------------------------------------------------------------- K4: hpT = p[j] * hT (bf16)
__global__ void k_scale_hp(const __bf16* __restrict__ hT, const float* __restrict__ p,
                           __bf16* __restrict__ hpT) {
  size_t idx = (size_t)blockIdx.x * blockDim.x + threadIdx.x;
  int j = (int)(idx & (N_NODES - 1));
  hpT[idx] = (__bf16)(p[j] * (float)hT[idx]);
}

// ---------------------------------------------------------------- K5: top half, masked half-sums via WMMA
__global__ void __launch_bounds__(256) k_attn_top(const int* __restrict__ supp,
                                                  const __bf16* __restrict__ hT,
                                                  const float* __restrict__ atop,
                                                  const float* __restrict__ btop,
                                                  float* __restrict__ out) {
  __shared__ __bf16 tileA[16 * 32];
  __shared__ float cnt[2][16];
  const int tid  = threadIdx.x;
  const int lane = tid & 31;
  const int wave = tid >> 5;
  const int hi   = lane >> 4;
  const int lcol = lane & 15;
  const int ibase = blockIdx.x * 16;

  Frag16 onescol; // B with ones in column 0 only -> accumulates row popcounts
#pragma unroll
  for (int e = 0; e < 16; ++e) onescol.h[e] = (lcol == 0) ? (__bf16)1.0f : (__bf16)0.0f;

  const int col0 = wave * 32 + lcol;
  const int col1 = col0 + 16;
  v8f accS[2][2] = {};
  v8f accC[2]    = {};

#pragma unroll
  for (int phase = 0; phase < 2; ++phase) {
    const int kstart = phase * HALF_N;
    for (int kb = kstart; kb < kstart + HALF_N; kb += 32) {
      __syncthreads();
      { // stage support tile -> bf16 0/1 mask in LDS
        int r  = tid >> 4;
        int c2 = (tid & 15) * 2;
        const int2 vv = *reinterpret_cast<const int2*>(supp + (size_t)(ibase + r) * N_NODES + kb + c2);
        tileA[r * 32 + c2]     = (vv.x > 0) ? (__bf16)1.0f : (__bf16)0.0f;
        tileA[r * 32 + c2 + 1] = (vv.y > 0) ? (__bf16)1.0f : (__bf16)0.0f;
      }
      __syncthreads();
      Frag16 a;
      {
        const __bf16* rp = &tileA[lcol * 32 + hi * 16];
        a.q[0] = *reinterpret_cast<const uint4*>(rp);
        a.q[1] = *reinterpret_cast<const uint4*>(rp + 8);
      }
      const int kc = kb + hi * 16;
      Frag16 b0, b1;
      { const uint4* pp = reinterpret_cast<const uint4*>(hT + (size_t)col0 * N_NODES + kc); b0.q[0]=pp[0]; b0.q[1]=pp[1]; }
      { const uint4* pp = reinterpret_cast<const uint4*>(hT + (size_t)col1 * N_NODES + kc); b1.q[0]=pp[0]; b1.q[1]=pp[1]; }
      accS[phase][0] = wmma_bf16(a.v, b0.v, accS[phase][0]);
      accS[phase][1] = wmma_bf16(a.v, b1.v, accS[phase][1]);
      if (wave == 0) accC[phase] = wmma_bf16(a.v, onescol.v, accC[phase]);
    }
  }
  __syncthreads();
  if (wave == 0 && lcol == 0) {
#pragma unroll
    for (int v = 0; v < 8; ++v) {
      cnt[0][hi * 8 + v] = accC[0][v];
      cnt[1][hi * 8 + v] = accC[1][v];
    }
  }
  __syncthreads();
#pragma unroll
  for (int v = 0; v < 8; ++v) {
    const int row = hi * 8 + v;
    const int i = ibase + row;
    const float la = atop[i], lb = btop[i];
    const float m  = fmaxf(la, lb);
    const float w1 = __expf(la - m), w2 = __expf(lb - m);
    const float inv = 1.0f / (w1 * cnt[0][row] + w2 * cnt[1][row]);
    out[(size_t)i * F_OUT + col0] = elu1((w1 * accS[0][0][v] + w2 * accS[1][0][v]) * inv);
    out[(size_t)i * F_OUT + col1] = elu1((w1 * accS[0][1][v] + w2 * accS[1][1][v]) * inv);
  }
}

// ---------------------------------------------------------------- K6: bottom half, mask @ (p*h) and mask @ p
__global__ void __launch_bounds__(256) k_attn_bot(const int* __restrict__ supp,
                                                  const __bf16* __restrict__ hpT,
                                                  const float* __restrict__ p,
                                                  float* __restrict__ out) {
  __shared__ __bf16 tileA[16 * 32];
  __shared__ float dens[16];
  const int tid  = threadIdx.x;
  const int lane = tid & 31;
  const int wave = tid >> 5;
  const int hi   = lane >> 4;
  const int lcol = lane & 15;
  const int ibase = HALF_N + blockIdx.x * 16;

  const int col0 = wave * 32 + lcol;
  const int col1 = col0 + 16;
  v8f accN0 = {}, accN1 = {}, accD = {};

  for (int kb = 0; kb < N_NODES; kb += 32) {
    __syncthreads();
    {
      int r  = tid >> 4;
      int c2 = (tid & 15) * 2;
      const int2 vv = *reinterpret_cast<const int2*>(supp + (size_t)(ibase + r) * N_NODES + kb + c2);
      tileA[r * 32 + c2]     = (vv.x > 0) ? (__bf16)1.0f : (__bf16)0.0f;
      tileA[r * 32 + c2 + 1] = (vv.y > 0) ? (__bf16)1.0f : (__bf16)0.0f;
    }
    __syncthreads();
    Frag16 a;
    {
      const __bf16* rp = &tileA[lcol * 32 + hi * 16];
      a.q[0] = *reinterpret_cast<const uint4*>(rp);
      a.q[1] = *reinterpret_cast<const uint4*>(rp + 8);
    }
    const int kc = kb + hi * 16;
    Frag16 b0, b1;
    { const uint4* pp = reinterpret_cast<const uint4*>(hpT + (size_t)col0 * N_NODES + kc); b0.q[0]=pp[0]; b0.q[1]=pp[1]; }
    { const uint4* pp = reinterpret_cast<const uint4*>(hpT + (size_t)col1 * N_NODES + kc); b1.q[0]=pp[0]; b1.q[1]=pp[1]; }
    accN0 = wmma_bf16(a.v, b0.v, accN0);
    accN1 = wmma_bf16(a.v, b1.v, accN1);
    if (wave == 0) { // denominator: B column 0 holds p[k]
      Frag16 pc;
      if (lcol == 0) {
#pragma unroll
        for (int e = 0; e < 16; ++e) pc.h[e] = (__bf16)p[kc + e];
      } else {
        pc.q[0] = make_uint4(0, 0, 0, 0);
        pc.q[1] = make_uint4(0, 0, 0, 0);
      }
      accD = wmma_bf16(a.v, pc.v, accD);
    }
  }
  __syncthreads();
  if (wave == 0 && lcol == 0) {
#pragma unroll
    for (int v = 0; v < 8; ++v) dens[hi * 8 + v] = accD[v];
  }
  __syncthreads();
#pragma unroll
  for (int v = 0; v < 8; ++v) {
    const int row = hi * 8 + v;
    const int i = ibase + row;
    const float inv = 1.0f / dens[row];
    out[(size_t)i * F_OUT + col0] = elu1(accN0[v] * inv);
    out[(size_t)i * F_OUT + col1] = elu1(accN1[v] * inv);
  }
}

// ---------------------------------------------------------------- launch
extern "C" void kernel_launch(void* const* d_in, const int* in_sizes, int n_in,
                              void* d_out, int out_size, void* d_ws, size_t ws_size,
                              hipStream_t stream) {
  const float* x    = (const float*)d_in[0];
  const int*   supp = (const int*)d_in[1];
  const float* W    = (const float*)d_in[2];
  const float* a    = (const float*)d_in[3];
  float* out = (float*)d_out;

  char* ws = (char*)d_ws;
  size_t off = 0;
  auto alloc = [&](size_t bytes) -> void* {
    void* ptr = ws + off;
    off = (off + bytes + 255) & ~(size_t)255;
    return ptr;
  };
  __bf16* hT  = (__bf16*)alloc((size_t)F_OUT * N_NODES * 2); // 4 MB, K-major h
  __bf16* hpT = (__bf16*)alloc((size_t)F_OUT * N_NODES * 2); // 4 MB, K-major p*h
  __bf16* WT  = (__bf16*)alloc((size_t)F_OUT * F_IN * 2);    // 256 KB
  float* s    = (float*)alloc((size_t)N_NODES * 4);
  float* t    = (float*)alloc((size_t)N_NODES * 4);
  float* p    = (float*)alloc((size_t)N_NODES * 4);
  float* atop = (float*)alloc((size_t)HALF_N * 4);
  float* btop = (float*)alloc((size_t)HALF_N * 4);

  k_transpose_w<<<(F_IN * F_OUT) / 256, 256, 0, stream>>>(W, WT);
  k_gemm_h<<<N_NODES / 16, 256, 0, stream>>>(x, WT, hT);
  k_st<<<N_NODES / 256, 256, 0, stream>>>(hT, a, s, t);
  k_scalars<<<1, 1024, 0, stream>>>(s, t, atop, btop, p);
  k_scale_hp<<<(F_OUT * N_NODES) / 256, 256, 0, stream>>>(hT, p, hpT);
  k_attn_top<<<HALF_N / 16, 256, 0, stream>>>(supp, hT, atop, btop, out);
  k_attn_bot<<<HALF_N / 16, 256, 0, stream>>>(supp, hpT, p, out);
}
